// JTNNDecoder_67207648248164
// MI455X (gfx1250) — compile-verified, matrix operand order checked by reading, not run on hardware
//
#include <hip/hip_runtime.h>
#include <hip/hip_bf16.h>
#include <cstdint>
#include <cstddef>

// Problem constants (from reference)
static constexpr int kB = 512;
static constexpr int kT = 64;
static constexpr int kH = 450;
static constexpr int kL = 56;
static constexpr int kV = 780;
static constexpr int kM = 8;
static constexpr int kS = kT * kB + 2;        // 32770 slots in h_buf
static constexpr int kR = (kT + 1) * kB;      // 33280 head rows

typedef __attribute__((ext_vector_type(16))) _Float16 v16h;
typedef __attribute__((ext_vector_type(8)))  float    v8f;

// ---------------------------------------------------------------------------
// WMMA GEMM: C[M,N] = act(A[M,K] (row-major, lda) * Wt[N,K]^T + bias)
// Block = 128 threads = 4 waves along N; block tile 64(M) x 64(N).
// Each wave owns a 16-wide N column group and FOUR 16x16 M-subtiles, so each
// K-chunk issues 4 back-to-back v_wmma_f32_16x16x32_f16 sharing one B frag.
// Tiles staged in LDS as f16 (convert once on stage), B stored transposed so
// fragment K-pairs merge into b32/b64 ds loads. All call sites have M%64==0.
// Interior K-chunks use a branch-free float2 staging fast path (8B-aligned:
// all lda*4 are multiples of 8 and k0*4 is a multiple of 128).
// act: 0=none 1=relu 2=sigmoid 3=tanh
// ---------------------------------------------------------------------------
__global__ __launch_bounds__(128)
void gemm_wmma_f16(const float* __restrict__ A, int lda,
                   const float* __restrict__ Wt, int ldw,
                   const float* __restrict__ bias,
                   float* __restrict__ C, int ldc,
                   int M, int N, int K, int act)
{
    __shared__ _Float16 As[64][36];   // [m][k] tile (f16, padded)
    __shared__ _Float16 Bt[64][36];   // [n][k] tile (f16, padded, transposed)

    const int wave = threadIdx.x >> 5;
    const int lane = threadIdx.x & 31;
    const int half = lane >> 4;
    const int l16  = lane & 15;

    const int m0 = blockIdx.y * 64;
    const int n0 = blockIdx.x * 64;
    const bool fullN = (n0 + 64 <= N);

    v8f acc0 = {}, acc1 = {}, acc2 = {}, acc3 = {};

    for (int k0 = 0; k0 < K; k0 += 32) {
        const bool fullK = (k0 + 32 <= K);

        // ---- Stage A tile (64 rows x 32 k) as f16 ----
        if (fullK) {
            // branch-free: 64*16 float2 loads across 128 threads (8 each)
            for (int q = threadIdx.x; q < 64 * 16; q += 128) {
                int r = q >> 4, c2 = (q & 15) << 1;
                float2 v = *(const float2*)&A[(size_t)(m0 + r) * lda + k0 + c2];
                As[r][c2]     = (_Float16)v.x;
                As[r][c2 + 1] = (_Float16)v.y;
            }
        } else {
            for (int q = threadIdx.x; q < 64 * 32; q += 128) {
                int r = q >> 5, c = q & 31;
                int gk = k0 + c;
                As[r][c] = (_Float16)((gk < K) ? A[(size_t)(m0 + r) * lda + gk] : 0.0f);
            }
        }
        // ---- Stage B tile (64 n x 32 k) transposed: Bt[n][k] = Wt[n0+n][k0+k] ----
        if (fullK && fullN) {
            for (int q = threadIdx.x; q < 64 * 16; q += 128) {
                int nn = q >> 4, c2 = (q & 15) << 1;
                float2 v = *(const float2*)&Wt[(size_t)(n0 + nn) * ldw + k0 + c2];
                Bt[nn][c2]     = (_Float16)v.x;
                Bt[nn][c2 + 1] = (_Float16)v.y;
            }
        } else {
            for (int q = threadIdx.x; q < 64 * 32; q += 128) {
                int nn = q >> 5, kk = q & 31;
                int gn = n0 + nn, gk = k0 + kk;
                Bt[nn][kk] = (_Float16)((gn < N && gk < K) ? Wt[(size_t)gn * ldw + gk] : 0.0f);
            }
        }
        // Prefetch next K-chunk into cache (gfx1250 global_prefetch_b8)
        if (k0 + 32 < K) {
            int r = threadIdx.x >> 5, c = threadIdx.x & 31;
            if (k0 + 32 + c < K)
                __builtin_prefetch(&A[(size_t)(m0 + r) * lda + k0 + 32 + c], 0, 3);
        }
        __syncthreads();

        // Fragment K index per documented 16-bit WMMA VGPR layout:
        // K(e) = ((e&8)<<1) + half*8 + (e&7); A row / B col = lane&15
        v16h bfrag;
#pragma unroll
        for (int e = 0; e < 16; ++e) {
            int kk = ((e & 8) << 1) + half * 8 + (e & 7);
            bfrag[e] = Bt[wave * 16 + l16][kk];
        }
        v16h a0, a1, a2, a3;
#pragma unroll
        for (int e = 0; e < 16; ++e) {
            int kk = ((e & 8) << 1) + half * 8 + (e & 7);
            a0[e] = As[ 0 + l16][kk];
            a1[e] = As[16 + l16][kk];
            a2[e] = As[32 + l16][kk];
            a3[e] = As[48 + l16][kk];
        }
        acc0 = __builtin_amdgcn_wmma_f32_16x16x32_f16(false, a0, false, bfrag,
                                                      (short)0, acc0, false, false);
        acc1 = __builtin_amdgcn_wmma_f32_16x16x32_f16(false, a1, false, bfrag,
                                                      (short)0, acc1, false, false);
        acc2 = __builtin_amdgcn_wmma_f32_16x16x32_f16(false, a2, false, bfrag,
                                                      (short)0, acc2, false, false);
        acc3 = __builtin_amdgcn_wmma_f32_16x16x32_f16(false, a3, false, bfrag,
                                                      (short)0, acc3, false, false);
        __syncthreads();
    }

    // Epilogue: acc_i[j] -> C[m0 + i*16 + half*8 + j][n0 + wave*16 + l16]
    const int col = n0 + wave * 16 + l16;
    if (col < N) {
        float bv = bias ? bias[col] : 0.0f;
        v8f accs[4] = {acc0, acc1, acc2, acc3};
#pragma unroll
        for (int i = 0; i < 4; ++i) {
#pragma unroll
            for (int j = 0; j < 8; ++j) {
                int row = m0 + i * 16 + half * 8 + j;
                float v = accs[i][j] + bv;
                if (act == 1)      v = fmaxf(v, 0.0f);
                else if (act == 2) v = 1.0f / (1.0f + __expf(-v));
                else if (act == 3) v = tanhf(v);
                C[(size_t)row * ldc + col] = v;
            }
        }
    }
}

// ---------------------------------------------------------------------------
// Utility / elementwise kernels
// ---------------------------------------------------------------------------
__global__ void zero_f32(float* p, size_t n) {
    size_t stride = (size_t)gridDim.x * blockDim.x;
    for (size_t i = (size_t)blockIdx.x * blockDim.x + threadIdx.x; i < n; i += stride)
        p[i] = 0.0f;
}

// Per-step gathers: xe, h_nei, sum_h, cur_o; build concat buffers.
__global__ __launch_bounds__(256)
void step_pre(int t, const float* __restrict__ emb, const int* __restrict__ x_wid,
              const int* __restrict__ h_idx, const int* __restrict__ o_idx,
              const float* __restrict__ h_buf,
              float* __restrict__ h_nei, float* __restrict__ sum_h,
              float* __restrict__ cur_o_all,
              float* __restrict__ cat_zs, float* __restrict__ cat_g)
{
    int i = blockIdx.x * 256 + threadIdx.x;
    if (i >= kB * kH) return;
    int b = i / kH, h = i % kH;
    int tb = t * kB + b;
    float xe = emb[(size_t)x_wid[tb] * kH + h];
    float sh = 0.0f;
#pragma unroll
    for (int m = 0; m < kM; ++m) {
        int idx = h_idx[(size_t)tb * kM + m];
        float hv = h_buf[(size_t)idx * kH + h];
        h_nei[((size_t)b * kM + m) * kH + h] = hv;
        sh += hv;
    }
    float co = 0.0f;
#pragma unroll
    for (int m = 0; m < kM; ++m)
        co += h_buf[(size_t)o_idx[(size_t)tb * kM + m] * kH + h];
    sum_h[i] = sh;
    cur_o_all[(size_t)tb * kH + h] = co;
    cat_zs[(size_t)b * (2 * kH) + h]      = xe;
    cat_zs[(size_t)b * (2 * kH) + kH + h] = sh;
    cat_g [(size_t)b * (2 * kH) + h]      = xe;
}

// r = sigmoid(xWr + nUr); gated = sum_m r * h_nei -> into cat_g second half
__global__ __launch_bounds__(256)
void step_rg(const float* __restrict__ xWr, const float* __restrict__ nUr,
             const float* __restrict__ h_nei, float* __restrict__ cat_g)
{
    int i = blockIdx.x * 256 + threadIdx.x;
    if (i >= kB * kH) return;
    int b = i / kH, h = i % kH;
    float xr = xWr[i];
    float g = 0.0f;
#pragma unroll
    for (int m = 0; m < kM; ++m) {
        size_t idx = ((size_t)b * kM + m) * kH + h;
        float r = 1.0f / (1.0f + __expf(-(xr + nUr[idx])));
        g += r * h_nei[idx];
    }
    cat_g[(size_t)b * (2 * kH) + kH + h] = g;
}

// new_h = (1-z)*sum_h + z*h_tilde; scatter into h_buf; record new_h_all
__global__ __launch_bounds__(256)
void step_fin(int t, const int* __restrict__ valid,
              const float* __restrict__ z, const float* __restrict__ sum_h,
              const float* __restrict__ htl,
              float* __restrict__ h_buf, float* __restrict__ new_h_all)
{
    int i = blockIdx.x * 256 + threadIdx.x;
    if (i >= kB * kH) return;
    int b = i / kH, h = i % kH;
    int tb = t * kB + b;
    float zz = z[i];
    float nh = (1.0f - zz) * sum_h[i] + zz * htl[i];
    new_h_all[(size_t)tb * kH + h] = nh;
    int v = valid[tb];
    int slot = (v > 0) ? (1 + tb) : (kS - 1);
    h_buf[(size_t)slot * kH + h] = (v > 0) ? nh : 0.0f;
}

__global__ __launch_bounds__(256)
void root_o_gather(const float* __restrict__ h_buf, const int* __restrict__ ridx,
                   float* __restrict__ root_o)
{
    int i = blockIdx.x * 256 + threadIdx.x;
    if (i >= kB * kH) return;
    int b = i / kH, h = i % kH;
    float s = 0.0f;
#pragma unroll
    for (int m = 0; m < kM; ++m)
        s += h_buf[(size_t)ridx[(size_t)b * kM + m] * kH + h];
    root_o[i] = s;
}

__global__ void build_pred_cat(const float* __restrict__ new_h_all,
                               const float* __restrict__ tv,
                               float* __restrict__ cat)
{
    const int W = kH + kL; // 506
    size_t n = (size_t)kR * W;
    size_t stride = (size_t)gridDim.x * blockDim.x;
    for (size_t i = (size_t)blockIdx.x * blockDim.x + threadIdx.x; i < n; i += stride) {
        int row = (int)(i / W), c = (int)(i % W);
        int b = (row < kB) ? row : ((row - kB) % kB);
        float v;
        if (c < kH) v = (row < kB) ? 0.0f : new_h_all[(size_t)(row - kB) * kH + c];
        else        v = tv[(size_t)b * kL + (c - kH)];
        cat[i] = v;
    }
}

__global__ void build_stop_cat(const float* __restrict__ emb,
                               const int* __restrict__ x_wid,
                               const int* __restrict__ root_wid,
                               const float* __restrict__ cur_o_all,
                               const float* __restrict__ root_o,
                               const float* __restrict__ tv,
                               float* __restrict__ cat)
{
    const int W = 2 * kH + kL; // 956
    size_t n = (size_t)kR * W;
    size_t stride = (size_t)gridDim.x * blockDim.x;
    for (size_t i = (size_t)blockIdx.x * blockDim.x + threadIdx.x; i < n; i += stride) {
        int row = (int)(i / W), c = (int)(i % W);
        float v;
        if (row < kT * kB) {
            int b = row % kB;
            if (c < kH)            v = emb[(size_t)x_wid[row] * kH + c];
            else if (c < 2 * kH)   v = cur_o_all[(size_t)row * kH + (c - kH)];
            else                   v = tv[(size_t)b * kL + (c - 2 * kH)];
        } else {
            int b = row - kT * kB;
            if (c < kH)            v = emb[(size_t)root_wid[b] * kH + c];
            else if (c < 2 * kH)   v = root_o[(size_t)b * kH + (c - kH)];
            else                   v = tv[(size_t)b * kL + (c - 2 * kH)];
        }
        cat[i] = v;
    }
}

// One wave32 per row: softmax CE + argmax over V=780, masked atomics.
__global__ __launch_bounds__(256)
void pred_ce(const float* __restrict__ scores, const int* __restrict__ root_wid,
             const int* __restrict__ pred_wid, const int* __restrict__ valid,
             const int* __restrict__ direction, float* __restrict__ acc, int rows)
{
    int row = blockIdx.x * 8 + (threadIdx.x >> 5);
    int lane = threadIdx.x & 31;
    if (row >= rows) return;
    const float* s = scores + (size_t)row * kV;
    float mask; int tgt;
    if (row < kB) { mask = 1.0f; tgt = root_wid[row]; }
    else {
        int r = row - kB;
        mask = (float)(valid[r] * direction[r]);
        tgt = pred_wid[r];
    }
    float mx = -1e30f; int amax = 0x7fffffff;
    for (int v = lane; v < kV; v += 32) {
        float x = s[v];
        if (x > mx) { mx = x; amax = v; }
    }
    for (int off = 16; off; off >>= 1) {
        float omx = __shfl_xor(mx, off);
        int   oax = __shfl_xor(amax, off);
        if (omx > mx || (omx == mx && oax < amax)) { mx = omx; amax = oax; }
    }
    float sum = 0.0f;
    for (int v = lane; v < kV; v += 32) sum += __expf(s[v] - mx);
    for (int off = 16; off; off >>= 1) sum += __shfl_xor(sum, off);
    if (lane == 0) {
        float ce = logf(sum) + mx - s[tgt];
        atomicAdd(&acc[0], ce * mask);
        atomicAdd(&acc[1], (amax == tgt ? 1.0f : 0.0f) * mask);
        atomicAdd(&acc[2], mask);
    }
}

// One wave32 per row: ss = dot(sv_row, Us) + bUs
__global__ __launch_bounds__(256)
void ss_dot(const float* __restrict__ sv, const float* __restrict__ Us,
            const float* __restrict__ bUs, float* __restrict__ ss, int rows)
{
    int row = blockIdx.x * 8 + (threadIdx.x >> 5);
    int lane = threadIdx.x & 31;
    if (row >= rows) return;
    float sum = 0.0f;
    for (int h = lane; h < kH; h += 32) sum += sv[(size_t)row * kH + h] * Us[h];
    for (int off = 16; off; off >>= 1) sum += __shfl_xor(sum, off);
    if (lane == 0) ss[row] = sum + bUs[0];
}

__global__ __launch_bounds__(256)
void stop_bce(const float* __restrict__ ss, const int* __restrict__ valid,
              const int* __restrict__ direction, float* __restrict__ acc, int rows)
{
    int row = blockIdx.x * 256 + threadIdx.x;
    if (row >= rows) return;
    float mask, tgt;
    if (row < kT * kB) { mask = (float)valid[row]; tgt = (float)direction[row]; }
    else               { mask = 1.0f; tgt = 0.0f; }
    float x = ss[row];
    float bce = fmaxf(x, 0.0f) - x * tgt + log1pf(__expf(-fabsf(x)));
    float stop = (x >= 0.0f) ? 1.0f : 0.0f;
    atomicAdd(&acc[3], bce * mask);
    atomicAdd(&acc[4], (stop == tgt ? 1.0f : 0.0f) * mask);
    atomicAdd(&acc[5], mask);
}

__global__ void finalize(const float* __restrict__ acc, float* __restrict__ out)
{
    if (threadIdx.x == 0 && blockIdx.x == 0) {
        out[0] = acc[0] / (float)kB;   // pred_loss
        out[1] = acc[3] / (float)kB;   // stop_loss
        out[2] = acc[1] / acc[2];      // pred_acc
        out[3] = acc[4] / acc[5];      // stop_acc
    }
}

// ---------------------------------------------------------------------------
// Host side
// ---------------------------------------------------------------------------
static inline void launch_gemm(const float* A, int lda, const float* Wt, int ldw,
                               const float* bias, float* C, int ldc,
                               int M, int N, int K, int act, hipStream_t s)
{
    dim3 g((N + 63) / 64, (M + 63) / 64);
    gemm_wmma_f16<<<g, dim3(128), 0, s>>>(A, lda, Wt, ldw, bias, C, ldc, M, N, K, act);
}

extern "C" void kernel_launch(void* const* d_in, const int* in_sizes, int n_in,
                              void* d_out, int out_size, void* d_ws, size_t ws_size,
                              hipStream_t stream)
{
    (void)in_sizes; (void)n_in; (void)out_size; (void)ws_size;
    const float* tree_vecs = (const float*)d_in[0];
    const int*   x_wid     = (const int*)d_in[1];
    const int*   pred_wid  = (const int*)d_in[2];
    const int*   root_wid  = (const int*)d_in[3];
    const int*   h_nei_idx = (const int*)d_in[4];
    const int*   o_nei_idx = (const int*)d_in[5];
    const int*   root_oidx = (const int*)d_in[6];
    const int*   valid     = (const int*)d_in[7];
    const int*   direction = (const int*)d_in[8];
    const float* emb = (const float*)d_in[9];
    const float* Wz  = (const float*)d_in[10];
    const float* bz  = (const float*)d_in[11];
    const float* Wr  = (const float*)d_in[12];
    const float* br  = (const float*)d_in[13];
    const float* Ur  = (const float*)d_in[14];
    const float* Wh  = (const float*)d_in[15];
    const float* bh  = (const float*)d_in[16];
    const float* Wm  = (const float*)d_in[17];
    const float* bW  = (const float*)d_in[18];
    const float* Um  = (const float*)d_in[19];
    const float* bU  = (const float*)d_in[20];
    const float* Wo  = (const float*)d_in[21];
    const float* bWo = (const float*)d_in[22];
    const float* Us  = (const float*)d_in[23];
    const float* bUs = (const float*)d_in[24];
    float* out = (float*)d_out;
    float* ws  = (float*)d_ws;

    // Workspace layout (floats). Persistent, then a phase-reused arena.
    // Peak ≈ 102M floats (~408 MB).
    size_t o_hbuf = 0;
    size_t o_newh = o_hbuf + (size_t)kS * kH;
    size_t o_curo = o_newh + (size_t)kT * kB * kH;
    size_t o_acc  = o_curo + (size_t)kT * kB * kH;
    size_t arena  = o_acc + 8;
    // scan-phase temporaries
    size_t o_hnei  = arena;
    size_t o_nur   = o_hnei  + (size_t)kB * kM * kH;
    size_t o_catzs = o_nur   + (size_t)kB * kM * kH;
    size_t o_catg  = o_catzs + (size_t)kB * 2 * kH;
    size_t o_sumh  = o_catg  + (size_t)kB * 2 * kH;
    size_t o_z     = o_sumh  + (size_t)kB * kH;
    size_t o_xwr   = o_z     + (size_t)kB * kH;
    size_t o_htl   = o_xwr   + (size_t)kB * kH;
    // pred-head phase (reuses arena after scan completes)
    size_t o_pcat = arena;
    size_t o_pv   = o_pcat + (size_t)kR * (kH + kL);
    size_t o_sc   = o_pv   + (size_t)kR * kH;
    // stop-head phase (reuses arena after pred CE completes)
    size_t o_scat = arena;
    size_t o_sv   = o_scat + (size_t)kR * (2 * kH + kL);
    size_t o_ss   = o_sv   + (size_t)kR * kH;
    size_t o_ro   = o_ss   + (size_t)kR;

    // Zero persistent state (deterministic across replays)
    zero_f32<<<2048, 256, 0, stream>>>(ws + o_hbuf, (size_t)kS * kH);
    zero_f32<<<1, 256, 0, stream>>>(ws + o_acc, 8);

    const int bhN = kB * kH;                  // 230400
    const int bhG = (bhN + 255) / 256;

    // -------- sequential GRU scan --------
    for (int t = 0; t < kT; ++t) {
        step_pre<<<bhG, 256, 0, stream>>>(t, emb, x_wid, h_nei_idx, o_nei_idx,
                                          ws + o_hbuf, ws + o_hnei, ws + o_sumh,
                                          ws + o_curo, ws + o_catzs, ws + o_catg);
        // z = sigmoid([x,sum_h] @ Wz^T + bz)
        launch_gemm(ws + o_catzs, 2 * kH, Wz, 2 * kH, bz, ws + o_z, kH,
                    kB, kH, 2 * kH, /*sigmoid*/2, stream);
        // xWr = x @ Wr^T + br   (A = first H cols of cat buffer)
        launch_gemm(ws + o_catzs, 2 * kH, Wr, kH, br, ws + o_xwr, kH,
                    kB, kH, kH, /*none*/0, stream);
        // nUr = h_nei @ Ur^T    [B*M, H]
        launch_gemm(ws + o_hnei, kH, Ur, kH, nullptr, ws + o_nur, kH,
                    kB * kM, kH, kH, /*none*/0, stream);
        step_rg<<<bhG, 256, 0, stream>>>(ws + o_xwr, ws + o_nur, ws + o_hnei,
                                         ws + o_catg);
        // h_tilde = tanh([x,gated] @ Wh^T + bh)
        launch_gemm(ws + o_catg, 2 * kH, Wh, 2 * kH, bh, ws + o_htl, kH,
                    kB, kH, 2 * kH, /*tanh*/3, stream);
        step_fin<<<bhG, 256, 0, stream>>>(t, valid, ws + o_z, ws + o_sumh,
                                          ws + o_htl, ws + o_hbuf, ws + o_newh);
    }

    // -------- pred (label) head --------
    build_pred_cat<<<2048, 256, 0, stream>>>(ws + o_newh, tree_vecs, ws + o_pcat);
    launch_gemm(ws + o_pcat, kH + kL, Wm, kH + kL, bW, ws + o_pv, kH,
                kR, kH, kH + kL, /*relu*/1, stream);
    launch_gemm(ws + o_pv, kH, Wo, kH, bWo, ws + o_sc, kV,
                kR, kV, kH, /*none*/0, stream);
    pred_ce<<<(kR + 7) / 8, 256, 0, stream>>>(ws + o_sc, root_wid, pred_wid,
                                              valid, direction, ws + o_acc, kR);

    // -------- stop (topological) head --------
    root_o_gather<<<bhG, 256, 0, stream>>>(ws + o_hbuf, root_oidx, ws + o_ro);
    build_stop_cat<<<4096, 256, 0, stream>>>(emb, x_wid, root_wid, ws + o_curo,
                                             ws + o_ro, tree_vecs, ws + o_scat);
    launch_gemm(ws + o_scat, 2 * kH + kL, Um, 2 * kH + kL, bU, ws + o_sv, kH,
                kR, kH, 2 * kH + kL, /*relu*/1, stream);
    ss_dot<<<(kR + 7) / 8, 256, 0, stream>>>(ws + o_sv, Us, bUs, ws + o_ss, kR);
    stop_bce<<<(kR + 255) / 256, 256, 0, stream>>>(ws + o_ss, valid, direction,
                                                   ws + o_acc, kR);

    finalize<<<1, 32, 0, stream>>>(ws + o_acc, out);
}
